// HawkBlock_85538568667437
// MI455X (gfx1250) — matrix operand order repaired and checked by reading, not compiled, not run on hardware
//
#include <hip/hip_runtime.h>
#include <hip/hip_bf16.h>
#include <stdint.h>

// Problem constants (match reference)
#define DMODEL 1024
#define DSTATE 1024
#define DFF    2048
#define BSZ    4
#define TSEQ   4096
#define BT     (BSZ * TSEQ)   // 16384 rows

// GEMM tiling: block = 256 threads (8 waves), block tile 128x256,
// wave tile 64x64 (4x4 WMMA tiles), K-step 32.
#define BLK_M 128
#define BLK_N 256
#define KSTEP 32
#define LDS_STRIDE 80                       // 64B of K data + 16B pad (conflict-minimal)
#define A_LDS_BYTES (BLK_M * LDS_STRIDE)    // 10240
#define B_LDS_BYTES (BLK_N * LDS_STRIDE)    // 20480
#define BUF_BYTES   (A_LDS_BYTES + B_LDS_BYTES)  // 30720 ; x2 buffers = 60 KB of 320 KB

typedef __attribute__((ext_vector_type(16))) __bf16 v16bf;
typedef __attribute__((ext_vector_type(8)))  float  v8f;
typedef int v4i __attribute__((__vector_size__(16)));  // matches builtin param type
typedef __attribute__((address_space(1))) v4i glb_v4i;
typedef __attribute__((address_space(3))) v4i lds_v4i;

// ---- CDNA5 async global->LDS path (guarded; sync fallback keeps compile green) ----
#if defined(__has_builtin)
#if __has_builtin(__builtin_amdgcn_global_load_async_to_lds_b128)
#define USE_ASYNC_LDS 1
#endif
#endif
#ifndef USE_ASYNC_LDS
#define USE_ASYNC_LDS 0
#endif

#if USE_ASYNC_LDS
#if __has_builtin(__builtin_amdgcn_s_wait_asynccnt)
#define WAIT_ASYNC0() __builtin_amdgcn_s_wait_asynccnt(0)
#else
#define WAIT_ASYNC0() asm volatile("s_wait_asynccnt 0x0" ::: "memory")
#endif
#else
#define WAIT_ASYNC0()
#endif

__device__ __forceinline__ uint16_t f2bf(float f) {
  uint32_t u = __float_as_uint(f);
  uint32_t r = (u + 0x7FFFu + ((u >> 16) & 1u)) >> 16;  // round-to-nearest-even
  return (uint16_t)r;
}
__device__ __forceinline__ float bf2f(uint16_t h) {
  return __uint_as_float(((uint32_t)h) << 16);
}

// ---------------- elementwise f32 -> bf16 (weights) ----------------
__global__ void cvt_bf16_kernel(const float* __restrict__ in,
                                uint16_t* __restrict__ out, int n) {
  int i = blockIdx.x * blockDim.x + threadIdx.x;
  if (i < n) out[i] = f2bf(in[i]);
}

// ---------------- LayerNorm over rows of DMODEL, write bf16 ----------------
__global__ void layernorm_kernel(const float* __restrict__ x,
                                 const float* __restrict__ g,
                                 const float* __restrict__ b,
                                 uint16_t* __restrict__ out) {
  __shared__ float s_sum[256];
  __shared__ float s_sq[256];
  const int row = blockIdx.x;
  const float* xr = x + (size_t)row * DMODEL;
  float4 v = ((const float4*)xr)[threadIdx.x];
  float sum = v.x + v.y + v.z + v.w;
  float sq  = v.x * v.x + v.y * v.y + v.z * v.z + v.w * v.w;
  s_sum[threadIdx.x] = sum;
  s_sq[threadIdx.x]  = sq;
  __syncthreads();
  for (int off = 128; off > 0; off >>= 1) {
    if (threadIdx.x < off) {
      s_sum[threadIdx.x] += s_sum[threadIdx.x + off];
      s_sq[threadIdx.x]  += s_sq[threadIdx.x + off];
    }
    __syncthreads();
  }
  const float mean = s_sum[0] * (1.0f / DMODEL);
  const float var  = s_sq[0] * (1.0f / DMODEL) - mean * mean;
  const float rstd = rsqrtf(var + 1e-5f);
  const int c = threadIdx.x * 4;
  uint16_t o0 = f2bf((v.x - mean) * rstd * g[c + 0] + b[c + 0]);
  uint16_t o1 = f2bf((v.y - mean) * rstd * g[c + 1] + b[c + 1]);
  uint16_t o2 = f2bf((v.z - mean) * rstd * g[c + 2] + b[c + 2]);
  uint16_t o3 = f2bf((v.w - mean) * rstd * g[c + 3] + b[c + 3]);
  uint2 packed;
  packed.x = (uint32_t)o0 | ((uint32_t)o1 << 16);
  packed.y = (uint32_t)o2 | ((uint32_t)o3 << 16);
  ((uint2*)(out + (size_t)row * DMODEL))[threadIdx.x] = packed;
}

// ---------------- LRU scan: one lane per (b, s) channel ----------------
__global__ void lru_scan_kernel(const uint16_t* __restrict__ u,
                                const float* __restrict__ log_lambda,
                                uint16_t* __restrict__ y) {
  const int tid = blockIdx.x * blockDim.x + threadIdx.x;  // 0 .. B*S-1
  const int b = tid >> 10;
  const int s = tid & (DSTATE - 1);
  const float lam = 1.0f / (1.0f + __expf(-log_lambda[s]));
  const float oml = 1.0f - lam;
  const uint16_t* up = u + (size_t)b * TSEQ * DSTATE + s;
  uint16_t*       yp = y + (size_t)b * TSEQ * DSTATE + s;
  float xs = 0.0f;
#pragma unroll 8
  for (int t = 0; t < TSEQ; ++t) {
    const float ut = bf2f(up[(size_t)t * DSTATE]);
    xs = lam * xs + oml * ut;
    yp[(size_t)t * DSTATE] = f2bf(xs);
  }
}

// ---------------- tile staging: global -> LDS (async DMA if available) ----------------
// A tile: BLK_M rows x 64 B of K data (512 x 16B chunks, 2 per thread)
// B tile: BLK_N rows x 64 B of K data (1024 x 16B chunks, 4 per thread)
// LDS rows padded to LDS_STRIDE so compute-side ds_load_b128 is conflict-minimal.
__device__ __forceinline__ void async_copy16(const void* g, void* l) {
#if USE_ASYNC_LDS
  __builtin_amdgcn_global_load_async_to_lds_b128(
      (glb_v4i*)(uintptr_t)g, (lds_v4i*)l, 0, 0);
#else
  *(uint4*)l = *(const uint4*)g;
#endif
}

__device__ __forceinline__ void stage_tile(const uint16_t* __restrict__ A,
                                           const uint16_t* __restrict__ W,
                                           int K, int rowBase, int colBase, int kt,
                                           uint8_t* sbuf, int tid) {
  uint8_t* sA = sbuf;
  uint8_t* sB = sbuf + A_LDS_BYTES;
#pragma unroll
  for (int q = 0; q < 2; ++q) {
    const int chunk = tid + q * 256;
    const int row = chunk >> 2, sub = chunk & 3;
    const uint8_t* g =
        (const uint8_t*)A + ((size_t)(rowBase + row) * K + kt) * 2 + sub * 16;
    async_copy16(g, sA + row * LDS_STRIDE + sub * 16);
  }
#pragma unroll
  for (int q = 0; q < 4; ++q) {
    const int chunk = tid + q * 256;
    const int col = chunk >> 2, sub = chunk & 3;
    const uint8_t* g =
        (const uint8_t*)W + ((size_t)(colBase + col) * K + kt) * 2 + sub * 16;
    async_copy16(g, sB + col * LDS_STRIDE + sub * 16);
  }
}

// ---------------- BF16 WMMA GEMM:  C[BT x N] = A[BT x K] * W[N x K]^T ----------------
// Epilogue MODE: 1 = store bf16 (u_proj)
//               2 = + D_skip[col]*u[row,col] + x[row,col] -> f32 out (x_mid)
//               3 = relu(acc + b1[col]) -> bf16 (h)
//               4 = out[row,col] += acc + b2[col]  (final residual, f32)
union FragBF {
  v16bf v;
  uint4 q[2];
};

template <int MODE>
__global__ __launch_bounds__(256) void gemm_wmma_kernel(
    const uint16_t* __restrict__ A,     // [BT x K] bf16 row-major
    const uint16_t* __restrict__ W,     // [N x K]  bf16 row-major
    const int K, const int N,
    float* __restrict__ outF,
    uint16_t* __restrict__ outH,
    const uint16_t* __restrict__ Ubf,
    const float* __restrict__ Xin,
    const float* __restrict__ Dskip,
    const float* __restrict__ bias) {
  __shared__ uint8_t smem[2 * BUF_BYTES];

  const int tid    = threadIdx.x;
  const int lane   = tid & 31;
  const int wave   = tid >> 5;   // 0..7
  const int laneLo = lane & 15;
  const int hi     = lane >> 4;  // 0 or 1
  const int mHalf  = wave >> 2;  // 0..1 -> 64-row half
  const int nQuad  = wave & 3;   // 0..3 -> 64-col quarter
  const int rowBase = blockIdx.x * BLK_M;
  const int colBase = blockIdx.y * BLK_N;

  v8f acc[4][4];
#pragma unroll
  for (int i = 0; i < 4; ++i)
#pragma unroll
    for (int j = 0; j < 4; ++j)
      acc[i][j] = (v8f){0.f, 0.f, 0.f, 0.f, 0.f, 0.f, 0.f, 0.f};

  const int steps = K / KSTEP;

  // prologue: stage first K-slice into buffer 0
  stage_tile(A, W, K, rowBase, colBase, 0, smem, tid);

  for (int it = 0; it < steps; ++it) {
    uint8_t* buf = smem + (size_t)(it & 1) * BUF_BYTES;
    WAIT_ASYNC0();        // our stage of buf has landed in LDS
    __syncthreads();      // everyone's stage landed; everyone done reading other buf
    if (it + 1 < steps) {
      stage_tile(A, W, K, rowBase, colBase, (it + 1) * KSTEP,
                 smem + (size_t)((it + 1) & 1) * BUF_BYTES, tid);
    }
    const uint8_t* sA = buf;
    const uint8_t* sB = buf + A_LDS_BYTES;

    FragBF af[4];
#pragma unroll
    for (int i = 0; i < 4; ++i) {
      // A 16-bit 16x32 frag: lane row M=laneLo, chunks at K bytes hi*16 and 32+hi*16
      const int m = mHalf * 64 + i * 16 + laneLo;
      af[i].q[0] = *(const uint4*)(sA + m * LDS_STRIDE + hi * 16);
      af[i].q[1] = *(const uint4*)(sA + m * LDS_STRIDE + 32 + hi * 16);
    }
    FragBF bfr[4];
#pragma unroll
    for (int j = 0; j < 4; ++j) {
      // B 16-bit 32x16 frag: lane col N=laneLo, chunks at K bytes hi*16 and 32+hi*16
      const int c = nQuad * 64 + j * 16 + laneLo;
      bfr[j].q[0] = *(const uint4*)(sB + c * LDS_STRIDE + hi * 16);
      bfr[j].q[1] = *(const uint4*)(sB + c * LDS_STRIDE + 32 + hi * 16);
    }
#pragma unroll
    for (int i = 0; i < 4; ++i)
#pragma unroll
      for (int j = 0; j < 4; ++j)
        acc[i][j] = __builtin_amdgcn_wmma_f32_16x16x32_bf16(
            false, af[i].v, false, bfr[j].v, (short)0, acc[i][j], false, false);
  }

  // Epilogue. C/D layout: VGPR r, lanes 0-15 -> M=r, lanes 16-31 -> M=8+r; N=laneLo.
#pragma unroll
  for (int i = 0; i < 4; ++i) {
#pragma unroll
    for (int j = 0; j < 4; ++j) {
      const int col = colBase + nQuad * 64 + j * 16 + laneLo;
#pragma unroll
      for (int r = 0; r < 8; ++r) {
        const int row = rowBase + mHalf * 64 + i * 16 + hi * 8 + r;
        const size_t idx = (size_t)row * N + col;
        float val = acc[i][j][r];
        if (MODE == 1) {
          outH[idx] = f2bf(val);
        } else if (MODE == 2) {
          outF[idx] = val + Dskip[col] * bf2f(Ubf[idx]) + Xin[idx];
        } else if (MODE == 3) {
          val += bias[col];
          outH[idx] = f2bf(val > 0.f ? val : 0.f);
        } else {  // MODE 4
          outF[idx] = outF[idx] + val + bias[col];
        }
      }
    }
  }
}

extern "C" void kernel_launch(void* const* d_in, const int* in_sizes, int n_in,
                              void* d_out, int out_size, void* d_ws, size_t ws_size,
                              hipStream_t stream) {
  const float* x          = (const float*)d_in[0];
  const float* w_in       = (const float*)d_in[1];
  const float* log_lambda = (const float*)d_in[2];
  const float* w_out      = (const float*)d_in[3];
  const float* D_skip     = (const float*)d_in[4];
  const float* ln1_g      = (const float*)d_in[5];
  const float* ln1_b      = (const float*)d_in[6];
  const float* ln2_g      = (const float*)d_in[7];
  const float* ln2_b      = (const float*)d_in[8];
  const float* w1         = (const float*)d_in[9];
  const float* b1         = (const float*)d_in[10];
  const float* w2         = (const float*)d_in[11];
  const float* b2         = (const float*)d_in[12];
  float* out = (float*)d_out;

  uint8_t* ws = (uint8_t*)d_ws;
  size_t off = 0;
  auto alloc = [&](size_t bytes) -> void* {
    void* p = ws + off;
    off += (bytes + 255) & ~(size_t)255;
    return p;
  };
  uint16_t* w_in_h  = (uint16_t*)alloc((size_t)DSTATE * DMODEL * 2);
  uint16_t* w_out_h = (uint16_t*)alloc((size_t)DMODEL * DSTATE * 2);
  uint16_t* w1_h    = (uint16_t*)alloc((size_t)DFF * DMODEL * 2);
  uint16_t* w2_h    = (uint16_t*)alloc((size_t)DMODEL * DFF * 2);
  uint16_t* ln1_h   = (uint16_t*)alloc((size_t)BT * DMODEL * 2);
  uint16_t* uproj_h = (uint16_t*)alloc((size_t)BT * DSTATE * 2);
  uint16_t* y_h     = (uint16_t*)alloc((size_t)BT * DSTATE * 2);
  uint16_t* h_h     = (uint16_t*)alloc((size_t)BT * DFF * 2);
  uint16_t* ln2_h   = uproj_h;  // u_proj dead after the scan

  {
    int n;
    n = DSTATE * DMODEL;
    cvt_bf16_kernel<<<(n + 255) / 256, 256, 0, stream>>>(w_in, w_in_h, n);
    n = DMODEL * DSTATE;
    cvt_bf16_kernel<<<(n + 255) / 256, 256, 0, stream>>>(w_out, w_out_h, n);
    n = DFF * DMODEL;
    cvt_bf16_kernel<<<(n + 255) / 256, 256, 0, stream>>>(w1, w1_h, n);
    n = DMODEL * DFF;
    cvt_bf16_kernel<<<(n + 255) / 256, 256, 0, stream>>>(w2, w2_h, n);
  }

  layernorm_kernel<<<BT, 256, 0, stream>>>(x, ln1_g, ln1_b, ln1_h);

  gemm_wmma_kernel<1><<<dim3(BT / BLK_M, DSTATE / BLK_N), 256, 0, stream>>>(
      ln1_h, w_in_h, DMODEL, DSTATE, nullptr, uproj_h, nullptr, nullptr, nullptr, nullptr);

  lru_scan_kernel<<<(BSZ * DSTATE) / 256, 256, 0, stream>>>(uproj_h, log_lambda, y_h);

  gemm_wmma_kernel<2><<<dim3(BT / BLK_M, DMODEL / BLK_N), 256, 0, stream>>>(
      y_h, w_out_h, DSTATE, DMODEL, out, nullptr, ln1_h, x, D_skip, nullptr);

  layernorm_kernel<<<BT, 256, 0, stream>>>(out, ln2_g, ln2_b, ln2_h);

  gemm_wmma_kernel<3><<<dim3(BT / BLK_M, DFF / BLK_N), 256, 0, stream>>>(
      ln2_h, w1_h, DMODEL, DFF, nullptr, h_h, nullptr, nullptr, nullptr, b1);

  gemm_wmma_kernel<4><<<dim3(BT / BLK_M, DMODEL / BLK_N), 256, 0, stream>>>(
      h_h, w2_h, DFF, DMODEL, out, nullptr, nullptr, nullptr, nullptr, b2);
}
